// LAINRDecoderGaussian_51359218925918
// MI455X (gfx1250) — matrix operand order repaired and checked by reading, not compiled, not run on hardware
//
#include <hip/hip_runtime.h>

typedef __attribute__((ext_vector_type(16))) _Float16 v16h;
typedef __attribute__((ext_vector_type(8)))  _Float16 v8h;
typedef __attribute__((ext_vector_type(8)))  float    v8f;

#define HW_TOT 16384

__device__ __forceinline__ v8f wmma_f16(v16h a, v16h b, v8f c) {
  return __builtin_amdgcn_wmma_f32_16x16x32_f16(false, a, false, b, (short)0, c, false, false);
}

// A fragment (16x32 f16, M x K) from row-major matrix, row stride ld, tile at (r0,k0).
// lane<16: M=lane, elems 0..15 = K k0..k0+15
// lane>=16: M=lane-16, elems 0..15 = K k0+16..k0+31
__device__ __forceinline__ v16h load_a16(const _Float16* p, int ld, int r0, int k0, int lane) {
  const int m = lane & 15;
  const int off = (lane >> 4) << 4;
  const _Float16* q = p + (size_t)(r0 + m) * ld + k0 + off;
  return *(const v16h*)q;
}

// B fragment (32x16 f16, K x N) where the source is stored transposed: wt[N][K] row-major.
// lane<16: holds B[k0..k0+15][n0+lane] ; lane>=16: B[k0+16..k0+31][n0+lane-16]
__device__ __forceinline__ v16h load_bT(const _Float16* wt, int ld, int n0, int k0, int lane) {
  const int n = lane & 15;
  const int off = (lane >> 4) << 4;
  return *(const v16h*)(wt + (size_t)(n0 + n) * ld + k0 + off);
}

// ---------------- prep kernels ----------------

__global__ void k_cvt_h(const float* __restrict__ s, _Float16* __restrict__ d, int n) {
  int i = blockIdx.x * 256 + threadIdx.x;
  if (i < n) d[i] = (_Float16)s[i];
}

__global__ void k_transpose_h(const float* __restrict__ src, _Float16* __restrict__ dst,
                              int R, int C) {
  int i = blockIdx.x * 256 + threadIdx.x;
  if (i < R * C) {
    int r = i / C, c = i % C;
    dst[(size_t)c * R + r] = (_Float16)src[i];
  }
}

__global__ void k_tarr(const float* __restrict__ coords, float* __restrict__ t_arr) {
  int i = blockIdx.x * 256 + threadIdx.x;
  if (i < HW_TOT) {
    float x = coords[2 * i], y = coords[2 * i + 1];
    int r = (int)(x * 16.0f); r = r < 0 ? 0 : (r > 15 ? 15 : r);
    int c = (int)(y * 16.0f); c = c < 0 ? 0 : (c > 15 ? 15 : c);
    t_arr[i] = (float)(r * 16 + c) * (1.0f / 256.0f);
  }
}

// ---------------- kv = tokens @ tokvW ----------------
// rows = b*256+tok (1024). K stored token-major [b][tok][128]; V stored transposed [b][h][d][tok].
__global__ void __launch_bounds__(128) k_kv(const _Float16* __restrict__ tok_h,
                                            const _Float16* __restrict__ tokvWt,
                                            _Float16* __restrict__ k_h,
                                            _Float16* __restrict__ vT_h) {
  const int lane = threadIdx.x & 31, wv = threadIdx.x >> 5;
  const int row0 = (blockIdx.x * 4 + wv) * 16;
  const int n = lane & 15, mo = (lane >> 4) << 3;
  v16h ta[8];
#pragma unroll
  for (int kb = 0; kb < 8; ++kb) ta[kb] = load_a16(tok_h, 256, row0, 32 * kb, lane);
#pragma unroll
  for (int t = 0; t < 16; ++t) {
    v8f c = {};
#pragma unroll
    for (int kb = 0; kb < 8; ++kb)
      c = wmma_f16(ta[kb], load_bT(tokvWt, 256, 16 * t, 32 * kb, lane), c);
    const int col = 16 * t + n;
#pragma unroll
    for (int j = 0; j < 8; ++j) {
      const int row = row0 + mo + j;
      const int b = row >> 8, tk = row & 255;
      if (col < 128) {
        k_h[(size_t)(b * 256 + tk) * 128 + col] = (_Float16)c[j];
      } else {
        const int cc = col - 128, hh = cc >> 6, d = cc & 63;
        vT_h[(size_t)((b * 2 + hh) * 64 + d) * 256 + tk] = (_Float16)c[j];
      }
    }
  }
}

// ---------------- q = relu(fourier(grid,Bq)@qW + qb) @ toqW ----------------
__global__ void __launch_bounds__(128) k_query(const float* __restrict__ coords,
                                               const float* __restrict__ Bq,
                                               const _Float16* __restrict__ qWt,
                                               const float* __restrict__ qb,
                                               const _Float16* __restrict__ toqWt,
                                               _Float16* __restrict__ q_h) {
  __shared__ __align__(32) _Float16 F[4][16 * 64];
  __shared__ __align__(32) _Float16 X[4][16 * 256];
  const int lane = threadIdx.x & 31, wv = threadIdx.x >> 5;
  const int row0 = (blockIdx.x * 4 + wv) * 16;
  const int n = lane & 15, mo = (lane >> 4) << 3;
  const float bx = Bq[2 * lane], by = Bq[2 * lane + 1];
#pragma unroll 4
  for (int r = 0; r < 16; ++r) {
    const float x = coords[2 * (row0 + r)], y = coords[2 * (row0 + r) + 1];
    const float pr = 6.283185307179586f * (x * bx + y * by);
    F[wv][r * 64 + lane]      = (_Float16)__cosf(pr);
    F[wv][r * 64 + 32 + lane] = (_Float16)__sinf(pr);
  }
  __syncthreads();
  v16h fa0 = load_a16(F[wv], 64, 0, 0, lane);
  v16h fa1 = load_a16(F[wv], 64, 0, 32, lane);
#pragma unroll
  for (int t = 0; t < 16; ++t) {
    v8f c = {};
    c = wmma_f16(fa0, load_bT(qWt, 64, 16 * t, 0, lane), c);
    c = wmma_f16(fa1, load_bT(qWt, 64, 16 * t, 32, lane), c);
    const int col = 16 * t + n;
    const float bias = qb[col];
#pragma unroll
    for (int j = 0; j < 8; ++j)
      X[wv][(mo + j) * 256 + col] = (_Float16)fmaxf(c[j] + bias, 0.0f);
  }
  __syncthreads();
  v16h xa[8];
#pragma unroll
  for (int kb = 0; kb < 8; ++kb) xa[kb] = load_a16(X[wv], 256, 0, 32 * kb, lane);
#pragma unroll
  for (int t = 0; t < 8; ++t) {
    v8f c = {};
#pragma unroll
    for (int kb = 0; kb < 8; ++kb)
      c = wmma_f16(xa[kb], load_bT(toqWt, 256, 16 * t, 32 * kb, lane), c);
#pragma unroll
    for (int j = 0; j < 8; ++j)
      q_h[(size_t)(row0 + mo + j) * 128 + 16 * t + n] = (_Float16)c[j];
  }
}

// ---------------- attention + mod projection ----------------
__global__ void __launch_bounds__(128) k_attn(const _Float16* __restrict__ q_h,
                                              const _Float16* __restrict__ k_h,
                                              const _Float16* __restrict__ vT_h,
                                              const _Float16* __restrict__ tooWt,
                                              const float* __restrict__ toob,
                                              const float* __restrict__ t_arr,
                                              _Float16* __restrict__ mod_h) {
  __shared__ __align__(32) _Float16 Ps[4][16 * 256];
  __shared__ __align__(32) _Float16 Os[4][16 * 128];
  const int lane = threadIdx.x & 31, wv = threadIdx.x >> 5;
  const int b = blockIdx.y;
  const int row0 = (blockIdx.x * 4 + wv) * 16;
  const int n = lane & 15, mo = (lane >> 4) << 3;
  const _Float16* Kb = k_h + (size_t)b * 256 * 128;
  const _Float16* Vb = vT_h + (size_t)b * 128 * 256;
  float trow[8];
#pragma unroll
  for (int j = 0; j < 8; ++j) trow[j] = t_arr[row0 + mo + j];

  for (int h = 0; h < 2; ++h) {
    v16h qa0 = load_a16(q_h, 128, row0, 64 * h, lane);
    v16h qa1 = load_a16(q_h, 128, row0, 64 * h + 32, lane);
    v8f S[16];
#pragma unroll
    for (int t = 0; t < 16; ++t) {
      v8f c = {};
      c = wmma_f16(qa0, load_bT(Kb, 128, 16 * t, 64 * h, lane), c);
      c = wmma_f16(qa1, load_bT(Kb, 128, 16 * t, 64 * h + 32, lane), c);
      const float pos = ((float)(16 * t + n) + 0.5f) * (1.0f / 256.0f);
#pragma unroll
      for (int j = 0; j < 8; ++j) {
        const float d = trow[j] - pos;
        S[t][j] = c[j] * 0.125f - 10.0f * d * d;
      }
    }
    // softmax over token axis (striped: 16 n-tiles in regs x 16 lanes per half)
    float sm[8];
#pragma unroll
    for (int j = 0; j < 8; ++j) {
      float m = S[0][j];
#pragma unroll
      for (int t = 1; t < 16; ++t) m = fmaxf(m, S[t][j]);
#pragma unroll
      for (int s = 1; s <= 8; s <<= 1) m = fmaxf(m, __shfl_xor(m, s, 32));
      float sum = 0.0f;
#pragma unroll
      for (int t = 0; t < 16; ++t) { float p = __expf(S[t][j] - m); S[t][j] = p; sum += p; }
#pragma unroll
      for (int s = 1; s <= 8; s <<= 1) sum += __shfl_xor(sum, s, 32);
      sm[j] = 1.0f / sum;
    }
#pragma unroll
    for (int t = 0; t < 16; ++t)
#pragma unroll
      for (int j = 0; j < 8; ++j)
        Ps[wv][(mo + j) * 256 + 16 * t + n] = (_Float16)(S[t][j] * sm[j]);
    __syncthreads();
    v16h pa[8];
#pragma unroll
    for (int kb = 0; kb < 8; ++kb) pa[kb] = load_a16(Ps[wv], 256, 0, 32 * kb, lane);
#pragma unroll
    for (int t = 0; t < 4; ++t) {
      v8f c = {};
#pragma unroll
      for (int kb = 0; kb < 8; ++kb)
        c = wmma_f16(pa[kb], load_bT(Vb, 256, 64 * h + 16 * t, 32 * kb, lane), c);
#pragma unroll
      for (int j = 0; j < 8; ++j)
        Os[wv][(mo + j) * 128 + 64 * h + 16 * t + n] = (_Float16)c[j];
    }
    __syncthreads();
  }
  // mod = O @ tooW + toob
  v16h ma[4];
#pragma unroll
  for (int kb = 0; kb < 4; ++kb) ma[kb] = load_a16(Os[wv], 128, 0, 32 * kb, lane);
#pragma unroll
  for (int t = 0; t < 16; ++t) {
    v8f c = {};
#pragma unroll
    for (int kb = 0; kb < 4; ++kb)
      c = wmma_f16(ma[kb], load_bT(tooWt, 128, 16 * t, 32 * kb, lane), c);
    const int col = 16 * t + n;
    const float bias = toob[col];
#pragma unroll
    for (int j = 0; j < 8; ++j)
      mod_h[((size_t)b * HW_TOT + row0 + mo + j) * 256 + col] = (_Float16)(c[j] + bias);
  }
}

// ---------------- bands + mod GEMMs + out0 partials, batch-pair B-reuse ----------------
// Each wave: one 16-row tile for batches (b0, b0+2). Band features/A-frags shared across the
// pair; every mod0t/mod1t B-fragment feeds 2 WMMAs. Writes a = m0+m1 (f16) and out0 part.
__global__ void __launch_bounds__(128) k_decode(
    const float* __restrict__ coords,
    const float* __restrict__ Bl0, const float* __restrict__ Bl1,
    const _Float16* __restrict__ bw0t, const float* __restrict__ bwb0,
    const _Float16* __restrict__ bw1t, const float* __restrict__ bwb1,
    const _Float16* __restrict__ mod0t, const float* __restrict__ modb0,
    const _Float16* __restrict__ mod1t, const float* __restrict__ modb1,
    const float* __restrict__ outW0, const float* __restrict__ outb0,
    const float* __restrict__ outb1,
    const _Float16* __restrict__ mod_h,
    _Float16* __restrict__ a_h,
    float* __restrict__ out) {
  __shared__ __align__(32) _Float16 F0[4][16 * 64];
  __shared__ __align__(32) _Float16 F1[4][16 * 64];
  const int lane = threadIdx.x & 31, wv = threadIdx.x >> 5;
  const int b0 = blockIdx.y;                       // batches b0 and b0+2
  const int row0 = (blockIdx.x * 4 + wv) * 16;
  const int n = lane & 15, mo = (lane >> 4) << 3;

  {
    const float b0x = Bl0[2 * lane], b0y = Bl0[2 * lane + 1];
    const float b1x = Bl1[2 * lane], b1y = Bl1[2 * lane + 1];
#pragma unroll 4
    for (int r = 0; r < 16; ++r) {
      const float x = coords[2 * (row0 + r)], y = coords[2 * (row0 + r) + 1];
      const float p0 = 6.283185307179586f * (x * b0x + y * b0y);
      const float p1 = 6.283185307179586f * (x * b1x + y * b1y);
      F0[wv][r * 64 + lane]      = (_Float16)__cosf(p0);
      F0[wv][r * 64 + 32 + lane] = (_Float16)__sinf(p0);
      F1[wv][r * 64 + lane]      = (_Float16)__cosf(p1);
      F1[wv][r * 64 + 32 + lane] = (_Float16)__sinf(p1);
    }
  }
  __syncthreads();
  v16h f0a = load_a16(F0[wv], 64, 0, 0, lane);
  v16h f0b = load_a16(F0[wv], 64, 0, 32, lane);
  v16h f1a = load_a16(F1[wv], 64, 0, 0, lane);
  v16h f1b = load_a16(F1[wv], 64, 0, 32, lane);
  v16h ma[2][8];
#pragma unroll
  for (int p = 0; p < 2; ++p)
#pragma unroll
    for (int kb = 0; kb < 8; ++kb)
      ma[p][kb] = load_a16(mod_h + (size_t)(b0 + 2 * p) * HW_TOT * 256, 256, row0, 32 * kb, lane);

  float ac0[2][8], ac1[2][8], ac2[2][8];
#pragma unroll
  for (int p = 0; p < 2; ++p)
#pragma unroll
    for (int j = 0; j < 8; ++j) { ac0[p][j] = 0.f; ac1[p][j] = 0.f; ac2[p][j] = 0.f; }

#pragma unroll 2
  for (int t = 0; t < 16; ++t) {
    v8f h0 = {}, h1 = {};
    h0 = wmma_f16(f0a, load_bT(bw0t, 64, 16 * t, 0, lane), h0);
    h0 = wmma_f16(f0b, load_bT(bw0t, 64, 16 * t, 32, lane), h0);
    h1 = wmma_f16(f1a, load_bT(bw1t, 64, 16 * t, 0, lane), h1);
    h1 = wmma_f16(f1b, load_bT(bw1t, 64, 16 * t, 32, lane), h1);
    v8f c0[2] = {{}, {}}, c1[2] = {{}, {}};
#pragma unroll
    for (int kb = 0; kb < 8; ++kb) {
      v16h bf0 = load_bT(mod0t, 256, 16 * t, 32 * kb, lane);
      c0[0] = wmma_f16(ma[0][kb], bf0, c0[0]);
      c0[1] = wmma_f16(ma[1][kb], bf0, c0[1]);
      v16h bf1 = load_bT(mod1t, 256, 16 * t, 32 * kb, lane);
      c1[0] = wmma_f16(ma[0][kb], bf1, c1[0]);
      c1[1] = wmma_f16(ma[1][kb], bf1, c1[1]);
    }
    const int col = 16 * t + n;
    const float bb0 = bwb0[col], bb1 = bwb1[col];
    const float mb0 = modb0[col], mb1 = modb1[col];
    const float w00 = outW0[col * 3 + 0], w01 = outW0[col * 3 + 1], w02 = outW0[col * 3 + 2];
#pragma unroll
    for (int p = 0; p < 2; ++p) {
#pragma unroll
      for (int j = 0; j < 8; ++j) {
        const float hh0 = fmaxf(h0[j] + bb0, 0.0f);
        const float m0  = fmaxf(hh0 + c0[p][j] + mb0, 0.0f);
        const float hh1 = fmaxf(h1[j] + bb1, 0.0f);
        const float m1  = fmaxf(hh1 + c1[p][j] + mb1, 0.0f);
        ac0[p][j] += m0 * w00; ac1[p][j] += m0 * w01; ac2[p][j] += m0 * w02;
        a_h[((size_t)(b0 + 2 * p) * HW_TOT + row0 + mo + j) * 256 + col] = (_Float16)(m0 + m1);
      }
    }
  }
  // reduce each row's 3 out0-partials across the 16 lanes of each half, write out0+biases
#pragma unroll
  for (int p = 0; p < 2; ++p)
#pragma unroll
    for (int j = 0; j < 8; ++j) {
#pragma unroll
      for (int s = 1; s <= 8; s <<= 1) {
        ac0[p][j] += __shfl_xor(ac0[p][j], s, 32);
        ac1[p][j] += __shfl_xor(ac1[p][j], s, 32);
        ac2[p][j] += __shfl_xor(ac2[p][j], s, 32);
      }
    }
  if (n < 3) {
    const float ob = outb0[n] + outb1[n];
#pragma unroll
    for (int p = 0; p < 2; ++p)
#pragma unroll
      for (int j = 0; j < 8; ++j) {
        const float v = (n == 0) ? ac0[p][j] : ((n == 1) ? ac1[p][j] : ac2[p][j]);
        out[((size_t)(b0 + 2 * p) * HW_TOT + row0 + mo + j) * 3 + n] = v + ob;
      }
  }
}

// ---------------- hv1 = relu(a @ hvW0 + hvb0); out += hv1 @ outW1 ----------------
// Two 16-row M-tiles per wave so every hv0t B-fragment feeds 2 WMMAs.
__global__ void __launch_bounds__(128) k_hv(const _Float16* __restrict__ a_h,
                                            const _Float16* __restrict__ hv0t,
                                            const float* __restrict__ hvb0,
                                            const float* __restrict__ outW1,
                                            float* __restrict__ out) {
  const int lane = threadIdx.x & 31, wv = threadIdx.x >> 5;
  const int row0 = (blockIdx.x * 4 + wv) * 32;     // over all 4*16384 rows
  const int n = lane & 15, mo = (lane >> 4) << 3;
  v16h aa[2][8];
#pragma unroll
  for (int p = 0; p < 2; ++p)
#pragma unroll
    for (int kb = 0; kb < 8; ++kb)
      aa[p][kb] = load_a16(a_h, 256, row0 + 16 * p, 32 * kb, lane);

  float ac0[2][8], ac1[2][8], ac2[2][8];
#pragma unroll
  for (int p = 0; p < 2; ++p)
#pragma unroll
    for (int j = 0; j < 8; ++j) { ac0[p][j] = 0.f; ac1[p][j] = 0.f; ac2[p][j] = 0.f; }

#pragma unroll 2
  for (int t = 0; t < 16; ++t) {
    v8f c[2] = {{}, {}};
#pragma unroll
    for (int kb = 0; kb < 8; ++kb) {
      v16h bf = load_bT(hv0t, 256, 16 * t, 32 * kb, lane);
      c[0] = wmma_f16(aa[0][kb], bf, c[0]);
      c[1] = wmma_f16(aa[1][kb], bf, c[1]);
    }
    const int col = 16 * t + n;
    const float hb = hvb0[col];
    const float w10 = outW1[col * 3 + 0], w11 = outW1[col * 3 + 1], w12 = outW1[col * 3 + 2];
#pragma unroll
    for (int p = 0; p < 2; ++p)
#pragma unroll
      for (int j = 0; j < 8; ++j) {
        const float hv1 = fmaxf(c[p][j] + hb, 0.0f);
        ac0[p][j] += hv1 * w10; ac1[p][j] += hv1 * w11; ac2[p][j] += hv1 * w12;
      }
  }
#pragma unroll
  for (int p = 0; p < 2; ++p)
#pragma unroll
    for (int j = 0; j < 8; ++j) {
#pragma unroll
      for (int s = 1; s <= 8; s <<= 1) {
        ac0[p][j] += __shfl_xor(ac0[p][j], s, 32);
        ac1[p][j] += __shfl_xor(ac1[p][j], s, 32);
        ac2[p][j] += __shfl_xor(ac2[p][j], s, 32);
      }
    }
  if (n < 3) {
#pragma unroll
    for (int p = 0; p < 2; ++p)
#pragma unroll
      for (int j = 0; j < 8; ++j) {
        const float v = (n == 0) ? ac0[p][j] : ((n == 1) ? ac1[p][j] : ac2[p][j]);
        const size_t idx = (size_t)(row0 + 16 * p + mo + j) * 3 + n;
        out[idx] = out[idx] + v;                   // k_decode wrote out0+biases first
      }
  }
}

// ---------------- host launcher ----------------

extern "C" void kernel_launch(void* const* d_in, const int* in_sizes, int n_in,
                              void* d_out, int out_size, void* d_ws, size_t ws_size,
                              hipStream_t stream) {
  const float* coords = (const float*)d_in[0];
  const float* tokens = (const float*)d_in[1];
  const float* B_q    = (const float*)d_in[2];
  const float* B_l0   = (const float*)d_in[3];
  const float* B_l1   = (const float*)d_in[4];
  const float* qW     = (const float*)d_in[5];
  const float* qb     = (const float*)d_in[6];
  const float* toqW   = (const float*)d_in[7];
  const float* tokvW  = (const float*)d_in[8];
  const float* tooW   = (const float*)d_in[9];
  const float* toob   = (const float*)d_in[10];
  const float* bwW0   = (const float*)d_in[11];
  const float* bwb0   = (const float*)d_in[12];
  const float* bwW1   = (const float*)d_in[13];
  const float* bwb1   = (const float*)d_in[14];
  const float* modW0  = (const float*)d_in[15];
  const float* modb0  = (const float*)d_in[16];
  const float* modW1  = (const float*)d_in[17];
  const float* modb1  = (const float*)d_in[18];
  const float* hvW0   = (const float*)d_in[19];
  const float* hvb0   = (const float*)d_in[20];
  const float* outW0  = (const float*)d_in[21];
  const float* outb0  = (const float*)d_in[22];
  const float* outW1  = (const float*)d_in[23];
  const float* outb1  = (const float*)d_in[24];

  char* w = (char*)d_ws;
  auto take = [&](size_t bytes) { char* p = w; w += (bytes + 255) & ~(size_t)255; return p; };
  _Float16* tokens_h = (_Float16*)take((size_t)4 * 256 * 256 * 2);
  _Float16* qWt      = (_Float16*)take((size_t)64 * 256 * 2);
  _Float16* toqWt    = (_Float16*)take((size_t)256 * 128 * 2);
  _Float16* tokvWt   = (_Float16*)take((size_t)256 * 256 * 2);
  _Float16* tooWt    = (_Float16*)take((size_t)128 * 256 * 2);
  _Float16* bw0t     = (_Float16*)take((size_t)64 * 256 * 2);
  _Float16* bw1t     = (_Float16*)take((size_t)64 * 256 * 2);
  _Float16* mod0t    = (_Float16*)take((size_t)256 * 256 * 2);
  _Float16* mod1t    = (_Float16*)take((size_t)256 * 256 * 2);
  _Float16* hv0t     = (_Float16*)take((size_t)256 * 256 * 2);
  _Float16* k_h      = (_Float16*)take((size_t)4 * 256 * 128 * 2);
  _Float16* vT_h     = (_Float16*)take((size_t)4 * 128 * 256 * 2);
  _Float16* q_h      = (_Float16*)take((size_t)HW_TOT * 128 * 2);
  float*    t_arr    = (float*)take((size_t)HW_TOT * 4);
  _Float16* mod_h    = (_Float16*)take((size_t)4 * HW_TOT * 256 * 2);
  _Float16* a_h      = (_Float16*)take((size_t)4 * HW_TOT * 256 * 2);

  k_cvt_h<<<dim3(1024), 256, 0, stream>>>(tokens, tokens_h, 4 * 256 * 256);
  k_transpose_h<<<dim3(64), 256, 0, stream>>>(qW, qWt, 64, 256);
  k_transpose_h<<<dim3(128), 256, 0, stream>>>(toqW, toqWt, 256, 128);
  k_transpose_h<<<dim3(256), 256, 0, stream>>>(tokvW, tokvWt, 256, 256);
  k_transpose_h<<<dim3(128), 256, 0, stream>>>(tooW, tooWt, 128, 256);
  k_transpose_h<<<dim3(64), 256, 0, stream>>>(bwW0, bw0t, 64, 256);
  k_transpose_h<<<dim3(64), 256, 0, stream>>>(bwW1, bw1t, 64, 256);
  k_transpose_h<<<dim3(256), 256, 0, stream>>>(modW0, mod0t, 256, 256);
  k_transpose_h<<<dim3(256), 256, 0, stream>>>(modW1, mod1t, 256, 256);
  k_transpose_h<<<dim3(256), 256, 0, stream>>>(hvW0, hv0t, 256, 256);
  k_tarr<<<dim3(64), 256, 0, stream>>>(coords, t_arr);

  k_kv<<<dim3(16), 128, 0, stream>>>(tokens_h, tokvWt, k_h, vT_h);
  k_query<<<dim3(256), 128, 0, stream>>>(coords, B_q, qWt, qb, toqWt, q_h);
  k_attn<<<dim3(256, 4), 128, 0, stream>>>(q_h, k_h, vT_h, tooWt, toob, t_arr, mod_h);
  k_decode<<<dim3(256, 2), 128, 0, stream>>>(coords, B_l0, B_l1,
      bw0t, bwb0, bw1t, bwb1, mod0t, modb0, mod1t, modb1,
      outW0, outb0, outb1, mod_h, a_h, (float*)d_out);
  k_hv<<<dim3(512), 128, 0, stream>>>(a_h, hv0t, hvb0, outW1, (float*)d_out);
}
